// LocalTransformerBlock1D_31181462569654
// MI455X (gfx1250) — matrix operand order repaired and checked by reading, not compiled, not run on hardware
//
#include <hip/hip_runtime.h>
#include <hip/hip_bf16.h>
#include <cstdint>

typedef unsigned short u16;
typedef unsigned int   u32;
typedef unsigned long long u64;

typedef __attribute__((ext_vector_type(8)))  float  v8f;
typedef __attribute__((ext_vector_type(8)))  u32    v8u;
typedef __attribute__((ext_vector_type(16))) __bf16 v16bf;
typedef __attribute__((ext_vector_type(4)))  u32    v4u;
typedef __attribute__((ext_vector_type(4)))  int    v4i;
typedef __attribute__((ext_vector_type(8)))  int    v8i;

#if __has_builtin(__builtin_amdgcn_tensor_load_to_lds) && __has_builtin(__builtin_amdgcn_s_wait_tensorcnt)
#define USE_TDM 1
#endif

// ---------------------------------------------------------------- helpers
__device__ inline u16 f2bf(float f) {
    u32 u = __float_as_uint(f);
    u += 0x7fffu + ((u >> 16) & 1u);   // round-to-nearest-even
    return (u16)(u >> 16);
}
__device__ inline float bf2f(u16 h) { return __uint_as_float(((u32)h) << 16); }
__device__ inline u32 pack2bf(float a, float b) {
    return (u32)f2bf(a) | ((u32)f2bf(b) << 16);
}

__device__ inline v8f v8f_zero() {
    v8f t;
#pragma unroll
    for (int i = 0; i < 8; ++i) t[i] = 0.0f;
    return t;
}

__device__ inline v8f wmma_bf16(v8u a, v8u b, v8f c) {
    return __builtin_amdgcn_wmma_f32_16x16x32_bf16(
        false, __builtin_bit_cast(v16bf, a),
        false, __builtin_bit_cast(v16bf, b),
        (short)0, c, false, false);
}

// A fragment (16x32 bf16): lane&15 -> row, (lane>>4)*8 -> K offset.
__device__ inline v8u ldsFragA(const u16* s, int stride, int row0, int kbase, int lane) {
    const u16* r = s + (size_t)(row0 + (lane & 15)) * stride + kbase + ((lane >> 4) << 3);
    v8u f;
#pragma unroll
    for (int i = 0; i < 4; ++i) f[i] = *(const u32*)(r + 2 * i);
#pragma unroll
    for (int i = 0; i < 4; ++i) f[4 + i] = *(const u32*)(r + 16 + 2 * i);
    return f;
}
// B operand (32x16 bf16) from LDS stored [N][K]: lane&15 -> column,
// lanes 0-15 hold K=0..7,16..23, lanes 16-31 hold K=8..15,24..31.
__device__ inline v8u ldsFragB(const u16* s, int stride, int col0, int kbase, int lane) {
    const u16* r = s + (size_t)(col0 + (lane & 15)) * stride + kbase + ((lane >> 4) << 3);
    v8u f;
#pragma unroll
    for (int i = 0; i < 4; ++i) f[i] = *(const u32*)(r + 2 * i);
#pragma unroll
    for (int i = 0; i < 4; ++i) f[4 + i] = *(const u32*)(r + 16 + 2 * i);
    return f;
}

// ---------------------------------------------------------------- convert (4 elems/thread)
__global__ void cvt_f32_bf16_v4(const float4* __restrict__ in, uint2* __restrict__ out, int n4) {
    int i = blockIdx.x * 256 + threadIdx.x;
    if (i < n4) {
        float4 f = in[i];
        uint2 o;
        o.x = pack2bf(f.x, f.y);
        o.y = pack2bf(f.z, f.w);
        out[i] = o;
    }
}

// ---------------------------------------------------------------- GEMM
// C[M,N] = A[M,K](bf16) * B[K,N](bf16)  (+bias), block tile 128x128, K step 32.
#define LDSK 36
template <bool OUTF32, bool HASBIAS>
__global__ __launch_bounds__(256) void gemm_bf16(
    const u16* __restrict__ A, const u16* __restrict__ B,
    void* __restrict__ Cv, const float* __restrict__ bias,
    int M, int N, int K, int lda, int ldb, int ldc)
{
    __shared__ u16 sA[128 * LDSK];
    __shared__ u16 sB[128 * LDSK];   // B^T tile: [n][k]
    const int tid  = threadIdx.x;
    const int lane = tid & 31;
    const int wid  = tid >> 5;
    const int wm   = (wid & 1) * 64;
    const int wn   = (wid >> 1) * 32;
    const int m0   = blockIdx.y * 128;
    const int n0   = blockIdx.x * 128;

    v8f acc[4][2];
#pragma unroll
    for (int i = 0; i < 4; ++i)
#pragma unroll
        for (int j = 0; j < 2; ++j) acc[i][j] = v8f_zero();

    for (int k0 = 0; k0 < K; k0 += 32) {
#ifdef USE_TDM
        // Tensor Data Mover: DMA the 128x32 A tile into padded LDS.
        // pad_interval=3 (16 DWORDs = one 32-elem row), pad_amount=1 (2 DWORDs)
        // reproduces the 36-element LDS row stride in hardware.
        if (wid == 0) {
            u64 ga  = (u64)(uintptr_t)(A + (size_t)m0 * lda + k0);
            u32 lds = (u32)(uintptr_t)&sA[0];
            v4u g0 = { 1u,                                   // count=1
                       lds,                                  // lds_addr
                       (u32)ga,                              // global_addr lo
                       (u32)((ga >> 32) & 0x01FFFFFFu) | (2u << 30) }; // hi + type=2
            u32 td0 = (u32)K, td1 = (u32)M;
            u64 st0 = (u64)lda;
            v8i g1 = {
                (int)((1u << 16) | (1u << 20) | (3u << 22) | (1u << 25)), // 2B, pad
                (int)((td0 & 0xFFFFu) << 16),
                (int)((td0 >> 16) | ((td1 & 0xFFFFu) << 16)),
                (int)((td1 >> 16) | (32u << 16)),            // tile_dim0=32
                (int)(128u),                                 // tile_dim1=128
                (int)(st0 & 0xFFFFFFFFu),
                (int)((st0 >> 32) & 0xFFFFu),
                0 };
            v4i gz = { 0, 0, 0, 0 };
#if __clang_major__ >= 23
            v8i gz8 = { 0, 0, 0, 0, 0, 0, 0, 0 };
            __builtin_amdgcn_tensor_load_to_lds(g0, g1, gz, gz, gz8, 0);
#else
            __builtin_amdgcn_tensor_load_to_lds(g0, g1, gz, gz, 0);
#endif
        }
#else
#pragma unroll
        for (int it = 0; it < 4; ++it) {               // A: 128x32
            int id  = tid + 256 * it;
            int row = id >> 3;
            int cc  = (id & 7) << 2;
            uint2 dv = *(const uint2*)(A + (size_t)(m0 + row) * lda + k0 + cc);
            *(uint2*)&sA[row * LDSK + cc] = dv;
        }
#endif
#pragma unroll
        for (int it = 0; it < 4; ++it) {               // B: 32x128 -> transposed
            int id = tid + 256 * it;
            int kk = id >> 5;
            int nc = (id & 31) << 2;
            uint2 dv = *(const uint2*)(B + (size_t)(k0 + kk) * ldb + n0 + nc);
            sB[(nc + 0) * LDSK + kk] = (u16)(dv.x);
            sB[(nc + 1) * LDSK + kk] = (u16)(dv.x >> 16);
            sB[(nc + 2) * LDSK + kk] = (u16)(dv.y);
            sB[(nc + 3) * LDSK + kk] = (u16)(dv.y >> 16);
        }
#ifdef USE_TDM
        if (wid == 0) __builtin_amdgcn_s_wait_tensorcnt(0);
#endif
        __syncthreads();
        v8u aF[4], bF[2];
#pragma unroll
        for (int i = 0; i < 4; ++i) aF[i] = ldsFragA(sA, LDSK, wm + 16 * i, 0, lane);
#pragma unroll
        for (int j = 0; j < 2; ++j) bF[j] = ldsFragB(sB, LDSK, wn + 16 * j, 0, lane);
#pragma unroll
        for (int i = 0; i < 4; ++i)
#pragma unroll
            for (int j = 0; j < 2; ++j)
                acc[i][j] = wmma_bf16(aF[i], bF[j], acc[i][j]);
        __syncthreads();
    }

    const int hi = lane >> 4, ln = lane & 15;
#pragma unroll
    for (int i = 0; i < 4; ++i)
#pragma unroll
        for (int j = 0; j < 2; ++j) {
            int col = n0 + wn + 16 * j + ln;
            float bv = 0.0f;
            if constexpr (HASBIAS) bv = bias[col];
            size_t base = (size_t)(m0 + wm + 16 * i + 8 * hi) * ldc + col;
#pragma unroll
            for (int r = 0; r < 8; ++r) {
                float v = acc[i][j][r] + bv;
                if constexpr (OUTF32) ((float*)Cv)[base + (size_t)r * ldc] = v;
                else                  ((u16*)Cv)[base + (size_t)r * ldc]  = f2bf(v);
            }
        }
}

// ---------------------------------------------------------------- RoPE (in place on q,k cols of qkv)
__global__ void rope_kernel(u16* __restrict__ qkv) {
    int idx = blockIdx.x * 256 + threadIdx.x;   // BT*256 threads
    int tok = idx >> 8;                          // b*T + t
    int rem = idx & 255;
    int h   = rem >> 5;
    int d2  = rem & 31;
    int t   = tok & 8191;
    float inv = __powf(10000.0f, -((float)(2 * d2)) / 64.0f);
    float s, c;
    __sincosf((float)t * inv, &s, &c);
    size_t base = (size_t)tok * 1536 + h * 64 + 2 * d2;
#pragma unroll
    for (int which = 0; which < 2; ++which) {     // 0: q, 1: k
        u16* p = qkv + base + which * 512;
        u32 u = *(u32*)p;
        float xe = bf2f((u16)u), xo = bf2f((u16)(u >> 16));
        *(u32*)p = pack2bf(xe * c - xo * s, xo * c + xe * s);
    }
}

// ---------------------------------------------------------------- local attention
// grid: (chunk 64, head 8, batch 4); 256 threads = 8 waves, 16 q-rows/wave.
__global__ __launch_bounds__(256) void attn_kernel(
    const u16* __restrict__ qkv, u16* __restrict__ ao)
{
    const int c = blockIdx.x, h = blockIdx.y, b = blockIdx.z;
    const int tid = threadIdx.x, lane = tid & 31, w = tid >> 5;
    const int hi = lane >> 4, ln = lane & 15;

    __shared__ u16 sQ[128 * 68];
    __shared__ u16 sK[64 * 68];
    __shared__ u16 sVT[64 * 68];
    __shared__ u16 sP[8 * 16 * 68];

    const size_t tokBase = (size_t)b * 8192;
    const int qcol = h * 64, kcol = 512 + h * 64, vcol = 1024 + h * 64;
    const int start = c * 128;

#pragma unroll
    for (int it = 0; it < 8; ++it) {               // Q 128x64
        int id  = tid + 256 * it;
        int row = id >> 4;
        int cc  = (id & 15) << 2;
        uint2 dv = *(const uint2*)(qkv + (tokBase + start + row) * 1536 + qcol + cc);
        *(uint2*)&sQ[row * 68 + cc] = dv;
    }
    __syncthreads();

    v8u qA0 = ldsFragA(sQ, 68, w * 16, 0, lane);
    v8u qA1 = ldsFragA(sQ, 68, w * 16, 32, lane);

    v8f oAcc[4];
#pragma unroll
    for (int n = 0; n < 4; ++n) oAcc[n] = v8f_zero();
    float rowM[8], rowL[8];
#pragma unroll
    for (int r = 0; r < 8; ++r) { rowM[r] = -3.0e38f; rowL[r] = 0.0f; }

    for (int kt = 0; kt < 6; ++kt) {
        const int j0 = kt * 64;
#pragma unroll
        for (int it = 0; it < 4; ++it) {           // K tile 64x64
            int id = tid + 256 * it;
            int kk = id >> 4;
            int cc = (id & 15) << 2;
            int kpos = start + j0 + kk - 127;
            uint2 dv = make_uint2(0u, 0u);
            if (kpos >= 0 && kpos < 8192)
                dv = *(const uint2*)(qkv + (tokBase + kpos) * 1536 + kcol + cc);
            *(uint2*)&sK[kk * 68 + cc] = dv;
        }
#pragma unroll
        for (int it = 0; it < 4; ++it) {           // V tile 64x64 -> transposed
            int id = tid + 256 * it;
            int kk = id >> 4;
            int cc = (id & 15) << 2;
            int kpos = start + j0 + kk - 127;
            uint2 dv = make_uint2(0u, 0u);
            if (kpos >= 0 && kpos < 8192)
                dv = *(const uint2*)(qkv + (tokBase + kpos) * 1536 + vcol + cc);
            sVT[(cc + 0) * 68 + kk] = (u16)(dv.x);
            sVT[(cc + 1) * 68 + kk] = (u16)(dv.x >> 16);
            sVT[(cc + 2) * 68 + kk] = (u16)(dv.y);
            sVT[(cc + 3) * 68 + kk] = (u16)(dv.y >> 16);
        }
        __syncthreads();

        // wave-uniform tile skip: rows [w*16, w*16+15] attend j in [i, i+255]
        const bool active = (j0 + 63 >= w * 16) && (j0 <= w * 16 + 15 + 255);
        if (active) {
            float sv[4][8];
#pragma unroll
            for (int n = 0; n < 4; ++n) {          // S = Q * K^T (16x64 per wave)
                v8u b0 = ldsFragB(sK, 68, 16 * n, 0, lane);
                v8u b1 = ldsFragB(sK, 68, 16 * n, 32, lane);
                v8f t = v8f_zero();
                t = wmma_bf16(qA0, b0, t);
                t = wmma_bf16(qA1, b1, t);
                int j = j0 + 16 * n + ln;
                int kpos = start + j - 127;
                bool kok = (kpos >= 0) && (kpos < 8192);
#pragma unroll
                for (int r = 0; r < 8; ++r) {
                    int i = w * 16 + r + 8 * hi;
                    bool ok = kok && (j >= i) && (j <= i + 255);
                    sv[n][r] = ok ? t[r] * 0.125f : -1.0e30f;
                }
            }
            // online softmax, rows reduced across the 16-lane half groups
#pragma unroll
            for (int r = 0; r < 8; ++r) {
                float m = fmaxf(fmaxf(sv[0][r], sv[1][r]), fmaxf(sv[2][r], sv[3][r]));
#pragma unroll
                for (int o = 1; o < 16; o <<= 1) m = fmaxf(m, __shfl_xor(m, o, 32));
                float nm = fmaxf(rowM[r], m);
                float corr = __expf(rowM[r] - nm);
                rowM[r] = nm;
                float ls = 0.0f;
#pragma unroll
                for (int n = 0; n < 4; ++n) {
                    float p = (sv[n][r] < -5.0e29f) ? 0.0f : __expf(sv[n][r] - nm);
                    sv[n][r] = p;
                    ls += p;
                }
#pragma unroll
                for (int o = 1; o < 16; o <<= 1) ls += __shfl_xor(ls, o, 32);
                rowL[r] = rowL[r] * corr + ls;
#pragma unroll
                for (int n = 0; n < 4; ++n) oAcc[n][r] *= corr;
            }
            // P -> LDS (per-wave region), rebuild A fragments, O += P * V
            u16* pb = sP + w * 16 * 68;
#pragma unroll
            for (int n = 0; n < 4; ++n)
#pragma unroll
                for (int r = 0; r < 8; ++r)
                    pb[(r + 8 * hi) * 68 + 16 * n + ln] = f2bf(sv[n][r]);
            v8u pA0 = ldsFragA(pb, 68, 0, 0, lane);
            v8u pA1 = ldsFragA(pb, 68, 0, 32, lane);
#pragma unroll
            for (int n = 0; n < 4; ++n) {
                v8u v0 = ldsFragB(sVT, 68, 16 * n, 0, lane);
                v8u v1 = ldsFragB(sVT, 68, 16 * n, 32, lane);
                oAcc[n] = wmma_bf16(pA0, v0, oAcc[n]);
                oAcc[n] = wmma_bf16(pA1, v1, oAcc[n]);
            }
        }
        __syncthreads();
    }
#pragma unroll
    for (int n = 0; n < 4; ++n) {
        int col = h * 64 + 16 * n + ln;
#pragma unroll
        for (int r = 0; r < 8; ++r) {
            int row = start + w * 16 + r + 8 * hi;
            ao[(tokBase + row) * 512 + col] = f2bf(oAcc[n][r] / rowL[r]);
        }
    }
}

// ---------------------------------------------------------------- SwiGLU (4 elems/thread, into gate buffer)
__global__ void swiglu_kernel(uint2* __restrict__ g, const uint2* __restrict__ v, int n4) {
    int i = blockIdx.x * 256 + threadIdx.x;
    if (i < n4) {
        uint2 gu = g[i], vu = v[i];
        float g0 = bf2f((u16)gu.x), g1 = bf2f((u16)(gu.x >> 16));
        float g2 = bf2f((u16)gu.y), g3 = bf2f((u16)(gu.y >> 16));
        float v0 = bf2f((u16)vu.x), v1 = bf2f((u16)(vu.x >> 16));
        float v2 = bf2f((u16)vu.y), v3 = bf2f((u16)(vu.y >> 16));
        uint2 o;
        o.x = pack2bf((g0 / (1.0f + __expf(-g0))) * v0, (g1 / (1.0f + __expf(-g1))) * v1);
        o.y = pack2bf((g2 / (1.0f + __expf(-g2))) * v2, (g3 / (1.0f + __expf(-g3))) * v3);
        g[i] = o;
    }
}

// ---------------------------------------------------------------- residual + RMSNorm (one block per row of 512)
__global__ __launch_bounds__(128) void norm_kernel(
    const float* __restrict__ xres, const void* __restrict__ add, int addF32,
    const float* __restrict__ scale, float* __restrict__ outF, u16* __restrict__ outB)
{
    const int row = blockIdx.x;
    const int tid = threadIdx.x;
    __shared__ float red[4];
    float y[4];
    float ss = 0.0f;
    size_t base = (size_t)row * 512;
#pragma unroll
    for (int i = 0; i < 4; ++i) {
        int col = tid + 128 * i;
        float a = addF32 ? ((const float*)add)[base + col]
                         : bf2f(((const u16*)add)[base + col]);
        float v = 2.4494897f * xres[base + col] + a;
        y[i] = v;
        ss += v * v;
    }
#pragma unroll
    for (int o = 1; o < 32; o <<= 1) ss += __shfl_xor(ss, o, 32);
    if ((tid & 31) == 0) red[tid >> 5] = ss;
    __syncthreads();
    float inv = rsqrtf((red[0] + red[1] + red[2] + red[3]) / 512.0f + 1.1920929e-7f);
#pragma unroll
    for (int i = 0; i < 4; ++i) {
        int col = tid + 128 * i;
        float v = y[i] * inv * scale[col];
        if (outF) outF[base + col] = v;
        if (outB) outB[base + col] = f2bf(v);
    }
}

// ---------------------------------------------------------------- host
extern "C" void kernel_launch(void* const* d_in, const int* in_sizes, int n_in,
                              void* d_out, int out_size, void* d_ws, size_t ws_size,
                              hipStream_t stream)
{
    (void)in_sizes; (void)n_in; (void)out_size;
    const int BT = 4 * 8192;

    const float* x    = (const float*)d_in[0];
    const float* Wqkv = (const float*)d_in[1];
    const float* outw = (const float*)d_in[2];
    const float* outb = (const float*)d_in[3];
    const float* n1s  = (const float*)d_in[4];
    const float* n2s  = (const float*)d_in[5];
    const float* ff1w = (const float*)d_in[6];
    const float* ff2w = (const float*)d_in[7];
    float* out = (float*)d_out;
    char* ws = (char*)d_ws;

    // workspace layout (bytes); phase-reused, peak = 377,487,360
    const size_t OFF_WB    = 0;            // bf16 weights, 8,388,608
    const size_t OFF_XB    = 8388608;      // bf16 x           33.5MB
    const size_t OFF_QKV   = 41943040;     // bf16 qkv        100.7MB
    const size_t OFF_AORAW = 142606336;    // bf16 attn raw    33.5MB
    const size_t OFF_AOP   = 176160768;    // bf16 attn proj   33.5MB
    const size_t OFF_X1F   = 8388608;      // f32 x1 (reuse xb/qkv head)
    const size_t OFF_X1B   = 75497472;     // bf16 x1
    const size_t OFF_GATE  = 109051904;    // bf16 gate/h     134.2MB
    const size_t OFF_VAL   = 243269632;    // bf16 val        134.2MB
    const size_t OFF_FF2   = 243269632;    // f32 ff2 out (reuses dead val)
    if (ws_size < 377487360ull) return;

    u16* wb    = (u16*)(ws + OFF_WB);
    u16* wqkvb = wb;
    u16* woutb = wb + 786432;
    u16* w1b   = wb + 1048576;
    u16* w2b   = wb + 3145728;
    u16* xb    = (u16*)(ws + OFF_XB);
    u16* qkvb  = (u16*)(ws + OFF_QKV);
    u16* aoraw = (u16*)(ws + OFF_AORAW);
    u16* aop   = (u16*)(ws + OFF_AOP);
    float* x1f = (float*)(ws + OFF_X1F);
    u16* x1b   = (u16*)(ws + OFF_X1B);
    u16* gate  = (u16*)(ws + OFF_GATE);
    u16* val   = (u16*)(ws + OFF_VAL);
    float* ff2o = (float*)(ws + OFF_FF2);

    // converts (all sizes multiple of 4)
    auto cvt = [&](const float* src, u16* dst, int n) {
        cvt_f32_bf16_v4<<<(n / 4 + 255) / 256, 256, 0, stream>>>(
            (const float4*)src, (uint2*)dst, n / 4);
    };
    cvt(x, xb, BT * 512);
    cvt(Wqkv, wqkvb, 512 * 1536);
    cvt(outw, woutb, 512 * 512);
    cvt(ff1w, w1b, 512 * 4096);
    cvt(ff2w, w2b, 2048 * 512);

    // qkv = x @ Wqkv
    gemm_bf16<false, false><<<dim3(12, 256), 256, 0, stream>>>(
        xb, wqkvb, qkvb, nullptr, BT, 1536, 512, 512, 1536, 1536);
    // RoPE on q,k
    rope_kernel<<<BT, 256, 0, stream>>>(qkvb);
    // local attention
    attn_kernel<<<dim3(64, 8, 4), 256, 0, stream>>>(qkvb, aoraw);
    // out projection + bias
    gemm_bf16<false, true><<<dim3(4, 256), 256, 0, stream>>>(
        aoraw, woutb, aop, outb, BT, 512, 512, 512, 512, 512);
    // x1 = rmsnorm(alpha*x + attn)
    norm_kernel<<<BT, 128, 0, stream>>>(x, aop, 0, n1s, x1f, x1b);
    // ff1 gate / val
    gemm_bf16<false, false><<<dim3(16, 256), 256, 0, stream>>>(
        x1b, w1b, gate, nullptr, BT, 2048, 512, 512, 4096, 2048);
    gemm_bf16<false, false><<<dim3(16, 256), 256, 0, stream>>>(
        x1b, w1b + 2048, val, nullptr, BT, 2048, 512, 512, 4096, 2048);
    // h = silu(gate)*val  (into gate buffer)
    swiglu_kernel<<<(BT * 2048 / 4 + 255) / 256, 256, 0, stream>>>(
        (uint2*)gate, (const uint2*)val, BT * 2048 / 4);
    // ff2
    gemm_bf16<true, false><<<dim3(4, 256), 256, 0, stream>>>(
        gate, w2b, ff2o, nullptr, BT, 512, 2048, 2048, 512, 512);
    // out = rmsnorm(alpha*x1 + h)
    norm_kernel<<<BT, 128, 0, stream>>>(x1f, ff2o, 1, n2s, out, nullptr);
}
